// CausalGatedAttention_5549097746792
// MI455X (gfx1250) — matrix operand order repaired and checked by reading, not compiled, not run on hardware
//
#include <hip/hip_runtime.h>
#include <hip/hip_bf16.h>

typedef __attribute__((ext_vector_type(16))) _Float16 v16h;
typedef __attribute__((ext_vector_type(8)))  _Float16 v8h;
typedef __attribute__((ext_vector_type(8)))  float    v8f;

#define EMBED 512
#define NHEAD 8
#define HDIM  64
#define BATCH 2
#define SEQ   4096
#define ROWS  (BATCH*SEQ)   /* 8192 */
#define QKV_N (3*EMBED)     /* 1536 */

// ---- WMMA operand loaders (ISA 7.12.2 layouts, wave32) ----

// A-matrix 16x32 f16: lane holds row (lane&15); 8 halves at k0+8*(lane>>4),
// 8 more at +16.
static __device__ __forceinline__ v16h load_a16(const _Float16* p, int ld,
                                                int row0, int k0, int lane) {
  const int r   = row0 + (lane & 15);
  const int klo = k0 + ((lane >> 4) << 3);
  const _Float16* base = p + (size_t)r * ld + klo;
  union { v16h v; v8h h[2]; } u;
  u.h[0] = *(const v8h*)(base);
  u.h[1] = *(const v8h*)(base + 16);
  return u.v;
}

// B-matrix 32x16 f16 where B[k][n] = W[n][k] (W row-major, leading dim ld):
// lane holds column (lane&15) = W row n; 8 k at k0+8*(lane>>4), 8 more at +16.
static __device__ __forceinline__ v16h load_bT(const _Float16* w, size_t ld,
                                               int n0, int k0, int lane) {
  const int n  = n0 + (lane & 15);
  const int kk = k0 + ((lane >> 4) << 3);
  const _Float16* base = w + (size_t)n * ld + kk;
  union { v16h v; v8h h[2]; } u;
  u.h[0] = *(const v8h*)(base);
  u.h[1] = *(const v8h*)(base + 16);
  return u.v;
}

static __device__ __forceinline__ v8f wmma16(v16h a, v16h b, v8f c) {
  return __builtin_amdgcn_wmma_f32_16x16x32_f16(false, a, false, b,
                                                (short)0, c, false, false);
}

// ---- Kernel 0: f32 -> f16 convert ----
__global__ void f32_to_f16(const float* __restrict__ src,
                           _Float16* __restrict__ dst, int n) {
  int i = blockIdx.x * blockDim.x + threadIdx.x;
  const int stride = gridDim.x * blockDim.x;
  for (; i < n; i += stride) dst[i] = (_Float16)src[i];
}

// ---- Kernel 1: QKV projection GEMM + head split (+ fold softmax scale into Q,
//      store V transposed per head). 32x64 tile per wave. ----
__global__ __launch_bounds__(128) void qkv_gemm(
    const _Float16* __restrict__ xh, const _Float16* __restrict__ wh,
    const float* __restrict__ bias,
    _Float16* __restrict__ qh, _Float16* __restrict__ kh,
    _Float16* __restrict__ vth) {
  const int lane = threadIdx.x & 31;
  const int wave = threadIdx.x >> 5;
  const int m0 = blockIdx.x * 128 + wave * 32;
  const int n0 = blockIdx.y * 64;
  v8f acc[2][4] = {};
  for (int k0 = 0; k0 < EMBED; k0 += 32) {
    v16h a0 = load_a16(xh, EMBED, m0,      k0, lane);
    v16h a1 = load_a16(xh, EMBED, m0 + 16, k0, lane);
#pragma unroll
    for (int t = 0; t < 4; ++t) {
      v16h b = load_bT(wh, EMBED, n0 + 16 * t, k0, lane);
      acc[0][t] = wmma16(a0, b, acc[0][t]);
      acc[1][t] = wmma16(a1, b, acc[1][t]);
    }
  }
  const int hi = lane >> 4, col = lane & 15;
#pragma unroll
  for (int t = 0; t < 4; ++t) {
    const int n   = n0 + 16 * t + col;
    const int sel = n / EMBED;            // 0=q, 1=k, 2=v
    const int h   = (n % EMBED) / HDIM;
    const int d   = n % HDIM;
    const float bn = bias[n];
#pragma unroll
    for (int mi = 0; mi < 2; ++mi) {
#pragma unroll
      for (int r = 0; r < 8; ++r) {
        const int m = m0 + 16 * mi + 8 * hi + r;
        const int bb = m / SEQ, s = m % SEQ;
        const float val = acc[mi][t][r] + bn;
        const size_t hoff = ((size_t)(bb * NHEAD + h) * SEQ + s) * HDIM + d;
        if (sel == 0)      qh[hoff] = (_Float16)(val * 0.125f);  // D^-0.5 = 1/8
        else if (sel == 1) kh[hoff] = (_Float16)val;
        else vth[((size_t)(bb * NHEAD + h) * HDIM + d) * SEQ + s] = (_Float16)val;
      }
    }
  }
}

// ---- Kernel 2: causal flash attention.
//      16 query rows per wave, 64-key tiles per online-softmax update. ----
__global__ __launch_bounds__(128) void attn_fa(
    const _Float16* __restrict__ qh, const _Float16* __restrict__ kh,
    const _Float16* __restrict__ vth, _Float16* __restrict__ attn) {
  __shared__ _Float16 lds_p[4][16][64];   // per-wave P staging (C- -> A-layout)
  const int lane = threadIdx.x & 31;
  const int wave = threadIdx.x >> 5;
  const int b = blockIdx.z, h = blockIdx.y;
  const int qbase = blockIdx.x * 64 + wave * 16;
  const _Float16* qp = qh  + (size_t)(b * NHEAD + h) * SEQ * HDIM;
  const _Float16* kp = kh  + (size_t)(b * NHEAD + h) * SEQ * HDIM;
  const _Float16* vp = vth + (size_t)(b * NHEAD + h) * HDIM * SEQ;
  const int hi = lane >> 4, col = lane & 15;

  const v16h aq0 = load_a16(qp, HDIM, qbase, 0,  lane);
  const v16h aq1 = load_a16(qp, HDIM, qbase, 32, lane);

  v8f o[4] = {};
  float mrow[8], lrow[8];
#pragma unroll
  for (int r = 0; r < 8; ++r) { mrow[r] = -1e30f; lrow[r] = 0.f; }

  const int qend = qbase + 15;
  for (int j0 = 0; j0 <= qend; j0 += 64) {   // causal: skip tiles past diagonal
    // ---- scores: 4 x (16q x 16k) tiles, d = 64 = 2 WMMA k-steps each ----
    v8f s[4];
#pragma unroll
    for (int u = 0; u < 4; ++u) {
      v8f z = {};
      v16h bk = load_bT(kp, HDIM, j0 + 16 * u, 0, lane);
      z = wmma16(aq0, bk, z);
      bk = load_bT(kp, HDIM, j0 + 16 * u, 32, lane);
      s[u] = wmma16(aq1, bk, z);
    }
    // ---- online softmax over 64 keys ----
#pragma unroll
    for (int r = 0; r < 8; ++r) {
      const int row = qbase + 8 * hi + r;
      float v[4];
#pragma unroll
      for (int u = 0; u < 4; ++u)
        v[u] = ((j0 + 16 * u + col) <= row) ? s[u][r] : -1e30f;
      float tmax = fmaxf(fmaxf(v[0], v[1]), fmaxf(v[2], v[3]));
#pragma unroll
      for (int sh = 1; sh < 16; sh <<= 1)
        tmax = fmaxf(tmax, __shfl_xor(tmax, sh, 32));
      const float mnew  = fmaxf(mrow[r], tmax);
      const float alpha = __expf(mrow[r] - mnew);
      float p[4], ps = 0.f;
#pragma unroll
      for (int u = 0; u < 4; ++u) { p[u] = __expf(v[u] - mnew); ps += p[u]; }
#pragma unroll
      for (int sh = 1; sh < 16; sh <<= 1)
        ps += __shfl_xor(ps, sh, 32);
      lrow[r] = lrow[r] * alpha + ps;
      mrow[r] = mnew;
#pragma unroll
      for (int t = 0; t < 4; ++t) o[t][r] *= alpha;
      const int rl = 8 * hi + r;
#pragma unroll
      for (int u = 0; u < 4; ++u)
        lds_p[wave][rl][16 * u + col] = (_Float16)p[u];
    }
    asm volatile("s_wait_dscnt 0x0" ::: "memory");  // P stores visible for reload
    const v16h ap0 = load_a16(&lds_p[wave][0][0], 64, 0, 0,  lane);
    const v16h ap1 = load_a16(&lds_p[wave][0][0], 64, 0, 32, lane);
    // ---- O += P @ V : K = 64 keys = 2 WMMA k-steps, N = 64 dims = 4 tiles ----
#pragma unroll
    for (int t = 0; t < 4; ++t) {
      v16h bv = load_bT(vp, SEQ, 16 * t, j0, lane);       // B[key][d] = vT[d][key]
      o[t] = wmma16(ap0, bv, o[t]);
      bv = load_bT(vp, SEQ, 16 * t, j0 + 32, lane);
      o[t] = wmma16(ap1, bv, o[t]);
    }
    asm volatile("s_wait_dscnt 0x0" ::: "memory");  // P reloads done before overwrite
  }

  _Float16* op = attn + (size_t)b * SEQ * EMBED + (size_t)h * HDIM;
#pragma unroll
  for (int r = 0; r < 8; ++r) {
    const float inv = 1.0f / lrow[r];
    const int row = qbase + 8 * hi + r;
#pragma unroll
    for (int t = 0; t < 4; ++t)
      op[(size_t)row * EMBED + 16 * t + col] = (_Float16)(o[t][r] * inv);
  }
}

// ---- Kernel 3: output projection GEMM (f32 out). 32x64 tile per wave. ----
__global__ __launch_bounds__(128) void out_gemm(
    const _Float16* __restrict__ ah, const _Float16* __restrict__ wh,
    const float* __restrict__ bias, float* __restrict__ out) {
  const int lane = threadIdx.x & 31;
  const int wave = threadIdx.x >> 5;
  const int m0 = blockIdx.x * 128 + wave * 32;
  const int n0 = blockIdx.y * 64;
  v8f acc[2][4] = {};
  for (int k0 = 0; k0 < EMBED; k0 += 32) {
    v16h a0 = load_a16(ah, EMBED, m0,      k0, lane);
    v16h a1 = load_a16(ah, EMBED, m0 + 16, k0, lane);
#pragma unroll
    for (int t = 0; t < 4; ++t) {
      v16h b = load_bT(wh, EMBED, n0 + 16 * t, k0, lane);
      acc[0][t] = wmma16(a0, b, acc[0][t]);
      acc[1][t] = wmma16(a1, b, acc[1][t]);
    }
  }
  const int hi = lane >> 4, col = lane & 15;
#pragma unroll
  for (int t = 0; t < 4; ++t) {
    const int n = n0 + 16 * t + col;
    const float bn = bias[n];
#pragma unroll
    for (int mi = 0; mi < 2; ++mi) {
#pragma unroll
      for (int r = 0; r < 8; ++r) {
        const int m = m0 + 16 * mi + 8 * hi + r;
        out[(size_t)m * EMBED + n] = acc[mi][t][r] + bn;
      }
    }
  }
}

extern "C" void kernel_launch(void* const* d_in, const int* in_sizes, int n_in,
                              void* d_out, int out_size, void* d_ws, size_t ws_size,
                              hipStream_t stream) {
  const float* x     = (const float*)d_in[0];
  // d_in[1] is the causal mask: computed analytically, never read (saves 64 MB)
  const float* qkv_w = (const float*)d_in[2];
  const float* qkv_b = (const float*)d_in[3];
  const float* out_w = (const float*)d_in[4];
  const float* out_b = (const float*)d_in[5];
  float* out = (float*)d_out;

  _Float16* ws = (_Float16*)d_ws;
  const size_t NX  = (size_t)ROWS  * EMBED;               // 4,194,304
  const size_t NW1 = (size_t)QKV_N * EMBED;               //   786,432
  const size_t NW2 = (size_t)EMBED * EMBED;               //   262,144
  const size_t NH  = (size_t)BATCH * NHEAD * SEQ * HDIM;  // 4,194,304
  _Float16* xh   = ws;
  _Float16* w1h  = xh  + NX;
  _Float16* w2h  = w1h + NW1;
  _Float16* qhp  = w2h + NW2;
  _Float16* khp  = qhp + NH;
  _Float16* vtp  = khp + NH;
  _Float16* atth = vtp + NH;   // total ~44 MB of d_ws

  f32_to_f16<<<2048, 256, 0, stream>>>(x,     xh,  (int)NX);
  f32_to_f16<<<768,  256, 0, stream>>>(qkv_w, w1h, (int)NW1);
  f32_to_f16<<<256,  256, 0, stream>>>(out_w, w2h, (int)NW2);

  qkv_gemm<<<dim3(ROWS / 128, QKV_N / 64), 128, 0, stream>>>(xh, w1h, qkv_b,
                                                             qhp, khp, vtp);
  attn_fa<<<dim3(SEQ / 64, NHEAD, BATCH), 128, 0, stream>>>(qhp, khp, vtp, atth);
  out_gemm<<<dim3(ROWS / 128, EMBED / 64), 128, 0, stream>>>(atth, w2h, out_b, out);
}
